// Point2PointAttention_3882650436826
// MI455X (gfx1250) — matrix operand order repaired and checked by reading, not compiled, not run on hardware
//
#include <hip/hip_runtime.h>
#include <hip/hip_bf16.h>

typedef __bf16 bf16;
typedef __attribute__((ext_vector_type(16))) __bf16 v16bf;
typedef __attribute__((ext_vector_type(8)))  float  v8f;

#define DEVINL __device__ __forceinline__
#define USE_ASYNC_LDS 1

static constexpr int BB = 32;     // batch
static constexpr int CC = 64;     // channels
static constexpr int NN = 2048;   // points
static constexpr int HH = 128;    // hidden
static constexpr float NEG_SLOPE = 0.2f;

// ---------------------------------------------------------------------------
// WMMA helpers (CDNA5 16x16x32 bf16, f32 accumulate)
// ---------------------------------------------------------------------------
DEVINL v8f vzero8() {
    v8f z;
#pragma unroll
    for (int i = 0; i < 8; ++i) z[i] = 0.0f;
    return z;
}

// A fragment (16x32, M x K): lane holds row (lane&15); K chunks
// [kb8, kb8+8) and [16+kb8, 16+kb8+8) with kb8 = (lane>>4)*8.
DEVINL v16bf load_a_frag(const bf16* rowk, int lane) {
    const int kb8 = (lane >> 4) * 8;
    union { v16bf v; uint4 q[2]; } u;
    u.q[0] = *(const uint4*)(rowk + kb8);
    u.q[1] = *(const uint4*)(rowk + 16 + kb8);
    return u.v;
}

// B fragment (32x16, K x N): lane holds column (lane&15); K chunks
// [kb8, kb8+8) and [16+kb8, 16+kb8+8) with kb8 = (lane>>4)*8.  Storage is B^T row-major.
DEVINL v16bf load_b_frag(const bf16* colk, int lane) {
    const int kb16 = (lane >> 4) * 8;
    union { v16bf v; uint4 q[2]; } u;
    u.q[0] = *(const uint4*)(colk + kb16);
    u.q[1] = *(const uint4*)(colk + 16 + kb16);
    return u.v;
}

DEVINL v8f wmma_bf16f32(v16bf a, v16bf b, v8f c) {
    return __builtin_amdgcn_wmma_f32_16x16x32_bf16(false, a, false, b,
                                                   (short)0, c, false, false);
}

// ---------------------------------------------------------------------------
// Async global->LDS copy (CDNA5 ASYNCcnt path), with sync fallback.
// ---------------------------------------------------------------------------
DEVINL unsigned lds_addr32(const void* p) {
    return (unsigned)(size_t)p;  // low 32 bits of LDS-aperture address
}

#if USE_ASYNC_LDS
DEVINL void async_copy_b128(unsigned lds_dst, const void* gbase, int voff) {
    asm volatile("global_load_async_to_lds_b128 %0, %1, %2 offset:0"
                 :: "v"(lds_dst), "v"(voff), "s"(gbase)
                 : "memory");
}
DEVINL void wait_async_all() {
    asm volatile("s_wait_asynccnt 0x0" ::: "memory");
}
#else
DEVINL void wait_async_all() {}
#endif

// ---------------------------------------------------------------------------
// Kernel 1: QKV projection.  x:[B,C,N] f32 -> q,k:[B,N,C] bf16 (q * 1/sqrt(C)),
// v:[B,C,N] bf16 (pre-transposed so attention streams it with b128 copies).
// ---------------------------------------------------------------------------
__global__ __launch_bounds__(128)
void qkv_proj_kernel(const float* __restrict__ x,
                     const float* __restrict__ wq,
                     const float* __restrict__ wk,
                     const float* __restrict__ wv,
                     bf16* __restrict__ qo, bf16* __restrict__ ko,
                     bf16* __restrict__ vo) {
    __shared__ __attribute__((aligned(16))) bf16 Xt[64 * 64];     // [n][c]
    __shared__ __attribute__((aligned(16))) bf16 Wl[3][64 * 64];  // [o][c]
    __shared__ __attribute__((aligned(16))) bf16 Vst[64 * 64];    // [c][n]
    const int tid = threadIdx.x;
    const int w = tid >> 5, lane = tid & 31;
    const int b = blockIdx.y, n0 = blockIdx.x * 64;

    for (int i = tid; i < 64 * 64; i += 128) {
        Wl[0][i] = (bf16)wq[i];
        Wl[1][i] = (bf16)wk[i];
        Wl[2][i] = (bf16)wv[i];
    }
    {   // load x tile, transpose into Xt[n][c]
        const int c = tid >> 1, hsel = tid & 1;
        const float* xs = x + ((size_t)b * CC + c) * NN + n0 + hsel * 32;
#pragma unroll
        for (int k = 0; k < 32; ++k) Xt[(hsel * 32 + k) * 64 + c] = (bf16)xs[k];
    }
    __syncthreads();

    const int row = lane & 15, half = lane >> 4, col = lane & 15;
    const v16bf xa0 = load_a_frag(&Xt[(w * 16 + row) * 64 + 0], lane);
    const v16bf xa1 = load_a_frag(&Xt[(w * 16 + row) * 64 + 32], lane);

    for (int m = 0; m < 3; ++m) {
        v8f acc[4];
#pragma unroll
        for (int ct = 0; ct < 4; ++ct) acc[ct] = vzero8();
#pragma unroll
        for (int ct = 0; ct < 4; ++ct) {
            v16bf b0 = load_b_frag(&Wl[m][(ct * 16 + col) * 64 + 0], lane);
            acc[ct] = wmma_bf16f32(xa0, b0, acc[ct]);
            v16bf b1 = load_b_frag(&Wl[m][(ct * 16 + col) * 64 + 32], lane);
            acc[ct] = wmma_bf16f32(xa1, b1, acc[ct]);
        }
        if (m < 2) {
            const float scale = (m == 0) ? 0.125f : 1.0f;  // 1/sqrt(64) for Q
            bf16* outp = (m == 0) ? qo : ko;
#pragma unroll
            for (int ct = 0; ct < 4; ++ct)
#pragma unroll
                for (int r = 0; r < 8; ++r) {
                    const int nn = n0 + w * 16 + r + 8 * half;
                    outp[((size_t)b * NN + nn) * CC + ct * 16 + col] =
                        (bf16)(acc[ct][r] * scale);
                }
        } else {
            // stage V transposed ([c][n]) then write [B,C,N] coalesced
#pragma unroll
            for (int ct = 0; ct < 4; ++ct)
#pragma unroll
                for (int r = 0; r < 8; ++r)
                    Vst[(ct * 16 + col) * 64 + w * 16 + r + 8 * half] =
                        (bf16)acc[ct][r];
            __syncthreads();
            const int c = tid >> 1, hsel = tid & 1;
            const uint4* src = (const uint4*)(&Vst[c * 64 + hsel * 32]);
            uint4* dst =
                (uint4*)(vo + ((size_t)b * CC + c) * NN + n0 + hsel * 32);
            dst[0] = src[0]; dst[1] = src[1]; dst[2] = src[2]; dst[3] = src[3];
        }
    }
}

// ---------------------------------------------------------------------------
// Kernel 2: flash attention.  q,k:[B,N,C] bf16, vt:[B,C,N] bf16 -> xr:[B,N,C]
// 4 waves x 16 query rows; 64-key chunks double-buffered in LDS via
// global_load_async_to_lds_b128; online softmax in f32.
// ---------------------------------------------------------------------------
__global__ __launch_bounds__(128)
void flash_attn_kernel(const bf16* __restrict__ q, const bf16* __restrict__ k,
                       const bf16* __restrict__ vt, float* __restrict__ xr) {
    __shared__ __attribute__((aligned(16))) bf16 Kl[2][64 * 64];  // [key][c]
    __shared__ __attribute__((aligned(16))) bf16 Vt[2][64 * 64];  // [c][key]
    __shared__ __attribute__((aligned(16))) bf16 Pl[4][16 * 64];  // per wave
    const int tid = threadIdx.x;
    const int w = tid >> 5, lane = tid & 31;
    const int b = blockIdx.y, n0 = blockIdx.x * 64;
    const int row = lane & 15, half = lane >> 4, col = lane & 15;

    const bf16* kbat  = k  + (size_t)b * NN * CC;  // K rows contiguous
    const bf16* vtbat = vt + (size_t)b * CC * NN;  // V^T rows stride NN

    const bf16* qrow = q + ((size_t)b * NN + n0 + w * 16 + row) * CC;
    const v16bf qa0 = load_a_frag(qrow + 0, lane);
    const v16bf qa1 = load_a_frag(qrow + 32, lane);

    v8f o[4];
#pragma unroll
    for (int i = 0; i < 4; ++i) o[i] = vzero8();
    float mrun[8], lrun[8];
#pragma unroll
    for (int r = 0; r < 8; ++r) { mrun[r] = -1e30f; lrun[r] = 0.0f; }

    // tile copy: K tile = contiguous 8KB; V^T tile = 64 rows of 128B
    auto issue_tile = [&](int buf, int key0) {
#if USE_ASYNC_LDS
        const bf16* kt0 = kbat + (size_t)key0 * CC;
        const bf16* vt0 = vtbat + key0;
#pragma unroll
        for (int i = 0; i < 4; ++i) {
            const int l = tid + 128 * i;                       // 0..511
            async_copy_b128(lds_addr32(&Kl[buf][0]) + l * 16, kt0, l * 16);
            const int vrow = l >> 3, vch = l & 7;
            async_copy_b128(lds_addr32(&Vt[buf][vrow * 64 + vch * 8]),
                            vt0, vrow * (NN * 2) + vch * 16);
        }
#else
        const int r2 = tid >> 1, hsel = tid & 1;
        const uint4* ks =
            (const uint4*)(kbat + ((size_t)key0 + r2) * CC + hsel * 32);
        uint4* kd = (uint4*)(&Kl[buf][r2 * 64 + hsel * 32]);
        kd[0] = ks[0]; kd[1] = ks[1]; kd[2] = ks[2]; kd[3] = ks[3];
        const uint4* vs =
            (const uint4*)(vtbat + (size_t)r2 * NN + key0 + hsel * 32);
        uint4* vd = (uint4*)(&Vt[buf][r2 * 64 + hsel * 32]);
        vd[0] = vs[0]; vd[1] = vs[1]; vd[2] = vs[2]; vd[3] = vs[3];
#endif
    };

    issue_tile(0, 0);  // prologue

    for (int j = 0; j < NN / 64; ++j) {
        const int cur = j & 1;
        wait_async_all();      // own async loads for `cur` have landed
        __syncthreads();       // => everyone's loads for `cur` have landed
        if (j + 1 < NN / 64) issue_tile(cur ^ 1, (j + 1) * 64);

        // S = Q * K^T  (16 x 64 per wave)
        v8f sacc[4];
#pragma unroll
        for (int i = 0; i < 4; ++i) sacc[i] = vzero8();
#pragma unroll
        for (int ct = 0; ct < 4; ++ct) {
            v16bf b0 = load_b_frag(&Kl[cur][(ct * 16 + col) * 64 + 0], lane);
            sacc[ct] = wmma_bf16f32(qa0, b0, sacc[ct]);
            v16bf b1 = load_b_frag(&Kl[cur][(ct * 16 + col) * 64 + 32], lane);
            sacc[ct] = wmma_bf16f32(qa1, b1, sacc[ct]);
        }

        // online softmax per row (rows live in 16-lane halves)
#pragma unroll
        for (int r = 0; r < 8; ++r) {
            float mx = fmaxf(fmaxf(sacc[0][r], sacc[1][r]),
                             fmaxf(sacc[2][r], sacc[3][r]));
#pragma unroll
            for (int d = 1; d < 16; d <<= 1) mx = fmaxf(mx, __shfl_xor(mx, d, 32));
            const float mnew = fmaxf(mrun[r], mx);
            const float corr = __expf(mrun[r] - mnew);
            float psum = 0.0f;
#pragma unroll
            for (int ct = 0; ct < 4; ++ct) {
                const float p = __expf(sacc[ct][r] - mnew);
                sacc[ct][r] = p;
                psum += p;
            }
#pragma unroll
            for (int d = 1; d < 16; d <<= 1) psum += __shfl_xor(psum, d, 32);
            lrun[r] = lrun[r] * corr + psum;
            mrun[r] = mnew;
#pragma unroll
            for (int ct = 0; ct < 4; ++ct) o[ct][r] *= corr;
        }

        // stage P (D-layout -> LDS row-major), then O += P * V
#pragma unroll
        for (int ct = 0; ct < 4; ++ct)
#pragma unroll
            for (int r = 0; r < 8; ++r)
                Pl[w][(r + 8 * half) * 64 + ct * 16 + col] = (bf16)sacc[ct][r];

#pragma unroll
        for (int kt = 0; kt < 2; ++kt) {
            v16bf pa = load_a_frag(&Pl[w][row * 64 + kt * 32], lane);
#pragma unroll
            for (int ct = 0; ct < 4; ++ct) {
                v16bf bv =
                    load_b_frag(&Vt[cur][(ct * 16 + col) * 64 + kt * 32], lane);
                o[ct] = wmma_bf16f32(pa, bv, o[ct]);
            }
        }
        __syncthreads();  // done reading `cur` before it is refilled
    }

#pragma unroll
    for (int r = 0; r < 8; ++r) {
        const float inv = 1.0f / lrun[r];
        const int nn = n0 + w * 16 + r + 8 * half;
        float* orow = xr + ((size_t)b * NN + nn) * CC;
#pragma unroll
        for (int ct = 0; ct < 4; ++ct) orow[ct * 16 + col] = o[ct][r] * inv;
    }
}

// ---------------------------------------------------------------------------
// Kernel 3: s = x + xr (transpose x via LDS) + BN1 channel sums.
// ---------------------------------------------------------------------------
__global__ __launch_bounds__(256)
void add_bn_stats_kernel(const float* __restrict__ x,
                         const float* __restrict__ xr,
                         float* __restrict__ s,
                         float* __restrict__ sum1, float* __restrict__ sq1) {
    __shared__ float xs[64 * 64];  // [c][n]
    __shared__ float chs[64], chs2[64];
    const int tid = threadIdx.x;
    const int b = blockIdx.y, n0 = blockIdx.x * 64;
    {
        const int c = tid >> 2, q4 = tid & 3;
        const float* src = x + ((size_t)b * CC + c) * NN + n0 + q4 * 16;
#pragma unroll
        for (int k = 0; k < 16; ++k) xs[c * 64 + q4 * 16 + k] = src[k];
    }
    if (tid < 64) { chs[tid] = 0.0f; chs2[tid] = 0.0f; }
    __syncthreads();

    const int n = tid >> 2, cq = tid & 3;
    const float* xrow = xr + ((size_t)b * NN + n0 + n) * CC;
    float* srow = s + ((size_t)b * NN + n0 + n) * CC;
#pragma unroll
    for (int k = 0; k < 16; ++k) {
        const int c = cq * 16 + k;
        const float val = xs[c * 64 + n] + xrow[c];
        srow[c] = val;
        atomicAdd(&chs[c], val);
        atomicAdd(&chs2[c], val * val);
    }
    __syncthreads();
    if (tid < 64) {
        atomicAdd(&sum1[tid], chs[tid]);
        atomicAdd(&sq1[tid], chs2[tid]);
    }
}

// ---------------------------------------------------------------------------
// BN finalize: a = gamma*rsqrt(var+eps), b = beta - mean*a
// ---------------------------------------------------------------------------
__global__ void bn_finalize_kernel(const float* __restrict__ sum,
                                   const float* __restrict__ sq,
                                   const float* __restrict__ gamma,
                                   const float* __restrict__ beta,
                                   float* __restrict__ a, float* __restrict__ bb) {
    const int c = threadIdx.x;
    const float inv_cnt = 1.0f / (float)((size_t)BB * NN);
    const float m = sum[c] * inv_cnt;
    const float var = sq[c] * inv_cnt - m * m;
    const float sc = gamma[c] * rsqrtf(var + 1e-5f);
    a[c] = sc;
    bb[c] = beta[c] - m * sc;
}

__global__ void zero_stats_kernel(float* __restrict__ p) {
    p[threadIdx.x] = 0.0f;  // 256 floats: sum1, sq1, sum2, sq2
}

// ---------------------------------------------------------------------------
// Kernel 5: h = BN1(s); ff = lrelu(W1 h); t = h + W2 ff; BN2 sums.
// ---------------------------------------------------------------------------
__global__ __launch_bounds__(128)
void ffn_kernel(const float* __restrict__ s,
                const float* __restrict__ w1, const float* __restrict__ w2,
                const float* __restrict__ a1, const float* __restrict__ b1,
                float* __restrict__ t,
                float* __restrict__ sum2, float* __restrict__ sq2) {
    __shared__ __attribute__((aligned(16))) bf16 W1l[HH * CC];   // [h][c]
    __shared__ __attribute__((aligned(16))) bf16 W2l[CC * HH];   // [c][h]
    __shared__ __attribute__((aligned(16))) bf16 hstage[4][16 * 64];
    __shared__ __attribute__((aligned(16))) bf16 fstage[4][16 * 128];
    __shared__ float hf[4][16 * 64];
    __shared__ float chs[64], chs2[64];
    const int tid = threadIdx.x;
    const int w = tid >> 5, lane = tid & 31;
    const int b = blockIdx.y, n0 = blockIdx.x * 64;
    const int row = lane & 15, half = lane >> 4, col = lane & 15;

    for (int i = tid; i < HH * CC; i += 128) W1l[i] = (bf16)w1[i];
    for (int i = tid; i < CC * HH; i += 128) W2l[i] = (bf16)w2[i];
    if (tid < 64) { chs[tid] = 0.0f; chs2[tid] = 0.0f; }

    {   // BN1 apply -> h (f32 + bf16 staged)
        const int n = n0 + w * 16 + row;
        const float* srow = s + ((size_t)b * NN + n) * CC;
        const int c0 = half * 32;
#pragma unroll
        for (int i = 0; i < 32; ++i) {
            const int c = c0 + i;
            const float hv = fmaf(srow[c], a1[c], b1[c]);
            hf[w][row * 64 + c] = hv;
            hstage[w][row * 64 + c] = (bf16)hv;
        }
    }
    __syncthreads();

    const v16bf ha0 = load_a_frag(&hstage[w][row * 64 + 0], lane);
    const v16bf ha1 = load_a_frag(&hstage[w][row * 64 + 32], lane);

    // FF1: 16 x 128
    v8f acc1[8];
#pragma unroll
    for (int i = 0; i < 8; ++i) acc1[i] = vzero8();
#pragma unroll
    for (int ct = 0; ct < 8; ++ct) {
        v16bf b0 = load_b_frag(&W1l[(ct * 16 + col) * 64 + 0], lane);
        acc1[ct] = wmma_bf16f32(ha0, b0, acc1[ct]);
        v16bf b1f = load_b_frag(&W1l[(ct * 16 + col) * 64 + 32], lane);
        acc1[ct] = wmma_bf16f32(ha1, b1f, acc1[ct]);
    }
    // LeakyReLU + stage as bf16 A operand
#pragma unroll
    for (int ct = 0; ct < 8; ++ct)
#pragma unroll
        for (int r = 0; r < 8; ++r) {
            float vv = acc1[ct][r];
            vv = (vv >= 0.0f) ? vv : NEG_SLOPE * vv;
            fstage[w][(r + 8 * half) * 128 + ct * 16 + col] = (bf16)vv;
        }

    // FF2: 16 x 64 over K = 128
    v8f acc2[4];
#pragma unroll
    for (int i = 0; i < 4; ++i) acc2[i] = vzero8();
#pragma unroll
    for (int kt = 0; kt < 4; ++kt) {
        v16bf fa = load_a_frag(&fstage[w][row * 128 + kt * 32], lane);
#pragma unroll
        for (int ct = 0; ct < 4; ++ct) {
            v16bf bf = load_b_frag(&W2l[(ct * 16 + col) * 128 + kt * 32], lane);
            acc2[ct] = wmma_bf16f32(fa, bf, acc2[ct]);
        }
    }

    // t = h + ff2, accumulate BN2 sums
#pragma unroll
    for (int ct = 0; ct < 4; ++ct)
#pragma unroll
        for (int r = 0; r < 8; ++r) {
            const int rr = r + 8 * half;
            const int cc = ct * 16 + col;
            const float tv = hf[w][rr * 64 + cc] + acc2[ct][r];
            t[((size_t)b * NN + n0 + w * 16 + rr) * CC + cc] = tv;
            atomicAdd(&chs[cc], tv);
            atomicAdd(&chs2[cc], tv * tv);
        }
    __syncthreads();
    if (tid < 64) {
        atomicAdd(&sum2[tid], chs[tid]);
        atomicAdd(&sq2[tid], chs2[tid]);
    }
}

// ---------------------------------------------------------------------------
// Kernel 7: out[b,c,n] = BN2(t[b,n,c]) with LDS-tiled transpose.
// ---------------------------------------------------------------------------
__global__ __launch_bounds__(256)
void bn_apply_transpose_kernel(const float* __restrict__ t,
                               const float* __restrict__ a2,
                               const float* __restrict__ b2,
                               float* __restrict__ out) {
    __shared__ float tile[64 * 65];  // padded
    const int tid = threadIdx.x;
    const int b = blockIdx.y, n0 = blockIdx.x * 64;
    {
        const int n = tid >> 2, cq = tid & 3;
        const float* trow = t + ((size_t)b * NN + n0 + n) * CC;
#pragma unroll
        for (int k = 0; k < 16; ++k) {
            const int c = cq * 16 + k;
            tile[n * 65 + c] = trow[c];
        }
    }
    __syncthreads();
    const int c = tid >> 2, nq = tid & 3;
    const float a = a2[c], bb = b2[c];
    float* orow = out + ((size_t)b * CC + c) * NN + n0 + nq * 16;
#pragma unroll
    for (int k = 0; k < 16; ++k) orow[k] = tile[(nq * 16 + k) * 65 + c] * a + bb;
}

// ---------------------------------------------------------------------------
// Host launch
// ---------------------------------------------------------------------------
extern "C" void kernel_launch(void* const* d_in, const int* in_sizes, int n_in,
                              void* d_out, int out_size, void* d_ws, size_t ws_size,
                              hipStream_t stream) {
    (void)in_sizes; (void)n_in; (void)out_size; (void)ws_size;
    const float* x   = (const float*)d_in[0];
    const float* wq  = (const float*)d_in[1];
    const float* wk  = (const float*)d_in[2];
    const float* wv  = (const float*)d_in[3];
    const float* w1  = (const float*)d_in[4];
    const float* w2  = (const float*)d_in[5];
    const float* g1  = (const float*)d_in[6];
    const float* be1 = (const float*)d_in[7];
    const float* g2  = (const float*)d_in[8];
    const float* be2 = (const float*)d_in[9];
    float* out = (float*)d_out;

    char* ws = (char*)d_ws;
    const size_t szQ = (size_t)BB * NN * CC * sizeof(bf16);   // 8 MB
    const size_t szF = (size_t)BB * NN * CC * sizeof(float);  // 16 MB
    bf16*  qbf  = (bf16*)(ws);
    bf16*  kbf  = (bf16*)(ws + szQ);
    bf16*  vbf  = (bf16*)(ws + 2 * szQ);      // [B,C,N] (transposed)
    float* xr   = (float*)(ws + 3 * szQ);
    float* sbuf = (float*)(ws);        // reuses q/k region (dead after attn)
    float* tbuf = xr;                  // reuses xr region (dead after stats)
    float* st   = (float*)(ws + 3 * szQ + szF);
    float *sum1 = st,       *sq1 = st + 64,  *sum2 = st + 128, *sq2 = st + 192;
    float *a1   = st + 256, *b1  = st + 320, *a2   = st + 384, *b2  = st + 448;

    dim3 grid(NN / 64, BB);
    zero_stats_kernel<<<1, 256, 0, stream>>>(st);
    qkv_proj_kernel<<<grid, 128, 0, stream>>>(x, wq, wk, wv, qbf, kbf, vbf);
    flash_attn_kernel<<<grid, 128, 0, stream>>>(qbf, kbf, vbf, xr);
    add_bn_stats_kernel<<<grid, 256, 0, stream>>>(x, xr, sbuf, sum1, sq1);
    bn_finalize_kernel<<<1, 64, 0, stream>>>(sum1, sq1, g1, be1, a1, b1);
    ffn_kernel<<<grid, 128, 0, stream>>>(sbuf, w1, w2, a1, b1, tbuf, sum2, sq2);
    bn_finalize_kernel<<<1, 64, 0, stream>>>(sum2, sq2, g2, be2, a2, b2);
    bn_apply_transpose_kernel<<<grid, 256, 0, stream>>>(tbuf, a2, b2, out);
}
